// EncoderDecoder_14173392077406
// MI455X (gfx1250) — compile-verified, matrix operand order checked by reading, not run on hardware
//
#include <hip/hip_runtime.h>
#include <hip/hip_bf16.h>
#include <cstdint>
#include <cstddef>

// ---------------------------------------------------------------------------
// MI455X (gfx1250) implementation.
//  - All GEMMs in bf16 via v_wmma_f32_16x16x32_bf16 (wave32, 16x16x32 tiles).
//  - bf16 weights (~150MB) are L2-resident (192MB L2) -> WMMA-bound, not HBM.
//  - Each wave computes 4 M-tiles x 1 N-tile: one B fragment feeds 4 WMMAs.
//  - 64x32 A-panel double-buffered in LDS, filled with gfx1250 async
//    global->LDS DMA (global_load_async_to_lds_b128 + s_wait_asynccnt),
//    so staging never round-trips through VGPRs or stalls on loadcnt.
//  - All 8 A-fragment ds_loads issued before the 4-WMMA chain so they
//    pipeline under a single dscnt wait.
// ---------------------------------------------------------------------------

typedef __attribute__((ext_vector_type(16))) __bf16 v16bf;
typedef __attribute__((ext_vector_type(8)))  float  v8f;

union Frag32 { uint4 q[2]; v16bf v; };

__device__ __forceinline__ uint16_t f2bf(float f) {
  union { float f; uint32_t u; } x; x.f = f;
  uint32_t r = x.u + 0x7FFFu + ((x.u >> 16) & 1u);   // RNE
  return (uint16_t)(r >> 16);
}
__device__ __forceinline__ float bf2f(uint16_t h) {
  union { uint32_t u; float f; } x; x.u = ((uint32_t)h) << 16;
  return x.f;
}

// Y[M,N] = act( X[M,K] * W[N,K]^T (+Y if ACC) + bias )
// flags: bit0 = accumulate into existing f32 Y, bit1 = ReLU.
// Y written at Y[row*ldy + col*csy]; Ybf (optional) gets a dense bf16 copy.
// Requirements (met by this model): M multiple of 64, N/16 multiple of 8,
// K multiple of 32.
__global__ __launch_bounds__(256) void gemm_bf16_wmma_kernel(
    const uint16_t* __restrict__ X, int ldx,
    const uint16_t* __restrict__ W, int ldw,
    const float* __restrict__ bias,
    float* __restrict__ Y, int ldy, int csy,
    uint16_t* __restrict__ Ybf, int ldybf,
    int M, int N, int K, int flags)
{
  __shared__ alignas(16) uint16_t ldsx[2][64 * 32];   // 2 x 4KB A panels

  const int lane = threadIdx.x & 31;
  const int wave = threadIdx.x >> 5;
  const int ntn  = N >> 4;
  const int ngrp = ntn >> 3;                     // 8 N-tiles per block
  const int mg   = blockIdx.x / ngrp;            // 64-row M group
  const int ng   = blockIdx.x - mg * ngrp;
  const int m0   = mg << 6;
  const int n0   = (ng * 8 + wave) << 4;
  const int lr   = lane & 15;
  const int hi   = lane >> 4;

  // Async-stage one 64x32 bf16 panel: 256 threads x one b128 (16B) each.
  const int sr = threadIdx.x >> 2;               // 0..63 panel row
  const int sc = (threadIdx.x & 3) << 3;         // 0,8,16,24 (bf16 elems)
  auto stage_async = [&](int buf, int k) {
    const uint16_t* gp = X + (size_t)(m0 + sr) * ldx + k + sc;
    // Low 32 bits of a flat pointer into LDS ARE the LDS byte address
    // (generic ptr = aperture_hi32 | lds_offset). Deriving the operand from
    // &ldsx[...] also makes the array escape into the asm, so the compiler
    // treats the DMA as a real store to ldsx (keeps the ds_loads below live).
    const uint32_t laddr = (uint32_t)(uintptr_t)&ldsx[buf][sr * 32 + sc];
    asm volatile("global_load_async_to_lds_b128 %0, %1, off"
                 :: "v"(laddr), "v"((unsigned long long)(uintptr_t)gp)
                 : "memory");
  };

  stage_async(0, 0);
  asm volatile("s_wait_asynccnt 0x0" ::: "memory");
  __syncthreads();

  v8f acc[4];
#pragma unroll
  for (int mt = 0; mt < 4; ++mt)
    acc[mt] = (v8f){0.f, 0.f, 0.f, 0.f, 0.f, 0.f, 0.f, 0.f};

  const uint16_t* Wrow = W + (size_t)(n0 + lr) * ldw;

  for (int k = 0; k < K; k += 32) {
    const int buf = (k >> 5) & 1;
    const bool more = (k + 32 < K);
    if (more) {
      stage_async(buf ^ 1, k + 32);              // DMA next panel, no VGPRs
      __builtin_prefetch(Wrow + k + 64, 0, 3);   // pull next W chunk nearer
    }
    // B fragment straight from [N,K] row-major weights (lane holds row
    // n0+lr, K elems {8*hi..+7} and {16+8*hi..+7}) -> X*W^T, no transpose.
    Frag32 fb;
    const uint4* pb = (const uint4*)(Wrow + k + hi * 8);
    fb.q[0] = pb[0];
    fb.q[1] = pb[2];
    // Issue ALL A-fragment ds_loads first (separate regs) so they pipeline
    // under one dscnt wait, then run the 4 WMMAs back-to-back.
    Frag32 fa[4];
#pragma unroll
    for (int mt = 0; mt < 4; ++mt) {
      const uint4* pa = (const uint4*)&ldsx[buf][(mt * 16 + lr) * 32 + hi * 8];
      fa[mt].q[0] = pa[0];
      fa[mt].q[1] = pa[2];
    }
#pragma unroll
    for (int mt = 0; mt < 4; ++mt)
      acc[mt] = __builtin_amdgcn_wmma_f32_16x16x32_bf16(
          false, fa[mt].v, false, fb.v, (short)0, acc[mt], false, false);
    if (more) asm volatile("s_wait_asynccnt 0x0" ::: "memory");
    __syncthreads();
  }

  // C/D layout: lane -> col n0+lr, VGPR j -> row base + 8*hi + j.
  const int   col = n0 + lr;
  const float bv  = bias ? bias[col] : 0.0f;
#pragma unroll
  for (int mt = 0; mt < 4; ++mt) {
#pragma unroll
    for (int j = 0; j < 8; ++j) {
      const int row = m0 + mt * 16 + hi * 8 + j;
      float v = acc[mt][j] + bv;
      if (flags & 1) v += Y[(size_t)row * ldy + (size_t)col * csy];
      if (flags & 2) v = fmaxf(v, 0.0f);
      if (Y)   Y[(size_t)row * ldy + (size_t)col * csy] = v;
      if (Ybf) Ybf[(size_t)row * ldybf + col] = f2bf(v);
    }
  }
}

// LSTM pointwise: G[B,4H] = preactivations (i,f,g,o). Updates c in place,
// writes new h as bf16 (GEMM input for the next step/layer); optionally also
// scatters h into the attention bank enc_bt[b, t, :] (extra pre-offset by t*H).
__global__ __launch_bounds__(256) void lstm_gates_kernel(
    const float* __restrict__ G, float* __restrict__ c,
    uint16_t* __restrict__ h_bf, uint16_t* __restrict__ extra,
    int H, int ML, int n)
{
  const int idx = blockIdx.x * blockDim.x + threadIdx.x;
  if (idx >= n) return;
  const int b = idx / H, j = idx - b * H;
  const float* g = G + (size_t)b * 4 * H;
  const float gi = g[j], gf = g[H + j], gg = g[2 * H + j], go = g[3 * H + j];
  const float si = 1.0f / (1.0f + __expf(-gi));
  const float sf = 1.0f / (1.0f + __expf(-gf));
  const float so = 1.0f / (1.0f + __expf(-go));
  const float c2 = sf * c[idx] + si * tanhf(gg);
  const float h2 = so * tanhf(c2);
  c[idx] = c2;
  h_bf[idx] = f2bf(h2);
  if (extra) extra[(size_t)b * ML * H + j] = f2bf(h2);
}

// Row softmax over [B, 128]: one wave per row (wave32 shuffles).
__global__ __launch_bounds__(256) void softmax128_kernel(
    const float* __restrict__ in, float* __restrict__ out, int B)
{
  const int wave = threadIdx.x >> 5, lane = threadIdx.x & 31;
  const int row = blockIdx.x * 8 + wave;
  if (row >= B) return;
  const float* r = in + (size_t)row * 128;
  float v[4], mx = -3.0e38f;
#pragma unroll
  for (int i = 0; i < 4; ++i) { v[i] = r[lane + 32 * i]; mx = fmaxf(mx, v[i]); }
  for (int off = 16; off; off >>= 1) mx = fmaxf(mx, __shfl_xor(mx, off, 32));
  float s = 0.f;
#pragma unroll
  for (int i = 0; i < 4; ++i) { v[i] = __expf(v[i] - mx); s += v[i]; }
  for (int off = 16; off; off >>= 1) s += __shfl_xor(s, off, 32);
  const float inv = 1.0f / s;
  float* o = out + (size_t)row * 128;
#pragma unroll
  for (int i = 0; i < 4; ++i) o[lane + 32 * i] = v[i] * inv;
}

// app[b,h] = sum_l aw[b,l] * enc_bt[b,l,h]   (33M MACs/step: VALU is fine;
// enc_bt bf16 (67MB) stays L2-resident across all 256 decoder steps).
__global__ __launch_bounds__(256) void attn_ctx_kernel(
    const float* __restrict__ aw, const uint16_t* __restrict__ encbt,
    uint16_t* __restrict__ app, int H, int ML, int n)
{
  const int idx = blockIdx.x * blockDim.x + threadIdx.x;
  if (idx >= n) return;
  const int b = idx / H, h = idx - b * H;
  const float* a = aw + (size_t)b * ML;
  const uint16_t* e = encbt + (size_t)b * ML * H + h;
  float s = 0.f;
  for (int l = 0; l < ML; ++l) s += a[l] * bf2f(e[(size_t)l * H]);
  app[idx] = f2bf(s);
}

__global__ __launch_bounds__(256) void cvt_f32_bf16_kernel(
    const float* __restrict__ in, uint16_t* __restrict__ out, int n)
{
  const int idx = blockIdx.x * blockDim.x + threadIdx.x;
  if (idx < n) out[idx] = f2bf(in[idx]);
}

__global__ __launch_bounds__(256) void fill_u32_kernel(uint32_t* p, int n)
{
  const int idx = blockIdx.x * blockDim.x + threadIdx.x;
  if (idx < n) p[idx] = 0u;
}

// ---------------------------------------------------------------------------
extern "C" void kernel_launch(void* const* d_in, const int* in_sizes, int n_in,
                              void* d_out, int out_size, void* d_ws, size_t ws_size,
                              hipStream_t stream)
{
  constexpr int S = 128, B = 256, IN = 256, H = 1024, ML = 128;
  (void)in_sizes; (void)n_in; (void)out_size; (void)ws_size;

  const float* input_data = (const float*)d_in[0];
  // d_in[1] = num_predictions (== S in the reference; compile-time here).
  int gi = 2;
  auto nxt = [&]() { return (const float*)d_in[gi++]; };

  const float *encW1 = nxt(), *encb1 = nxt(), *encW2 = nxt(), *encb2 = nxt();
  struct LWf { const float *Wih, *Whh, *bih, *bhh; };
  LWf encl[3];
  for (int l = 0; l < 3; ++l) { encl[l].Wih = nxt(); encl[l].Whh = nxt(); encl[l].bih = nxt(); encl[l].bhh = nxt(); }
  struct DecF { const float *We, *be, *Wa, *ba, *Wc, *bc; LWf l[3]; const float *mW[4], *mb[4]; };
  DecF df[2];
  for (int d = 0; d < 2; ++d) {
    DecF& D = df[d];
    D.We = nxt(); D.be = nxt(); D.Wa = nxt(); D.ba = nxt(); D.Wc = nxt(); D.bc = nxt();
    for (int l = 0; l < 3; ++l) { D.l[l].Wih = nxt(); D.l[l].Whh = nxt(); D.l[l].bih = nxt(); D.l[l].bhh = nxt(); }
    for (int m = 0; m < 4; ++m) { D.mW[m] = nxt(); D.mb[m] = nxt(); }
  }

  // -------- workspace carving (deterministic every call) --------
  uintptr_t cur = (uintptr_t)d_ws;
  auto alloc = [&](size_t bytes) -> void* {
    cur = (cur + 255) & ~(uintptr_t)255;
    void* r = (void*)cur; cur += bytes; return r;
  };
  auto tobf = [&](const float* src, size_t n) -> uint16_t* {
    uint16_t* dst = (uint16_t*)alloc(n * 2);
    cvt_f32_bf16_kernel<<<(int)((n + 255) / 256), 256, 0, stream>>>(src, dst, (int)n);
    return dst;
  };

  // bf16 weights (converted every call; ~150MB, lives in L2 while running)
  uint16_t* encW1b = tobf(encW1, (size_t)H * IN);
  uint16_t* encW2b = tobf(encW2, (size_t)IN * H);
  struct LWb { uint16_t *Wih, *Whh; };
  LWb enclb[3];
  for (int l = 0; l < 3; ++l) {
    enclb[l].Wih = tobf(encl[l].Wih, (size_t)4 * H * (l ? H : IN));
    enclb[l].Whh = tobf(encl[l].Whh, (size_t)4 * H * H);
  }
  struct DecB { uint16_t *We, *Wa, *Wc; LWb l[3]; uint16_t* mW[4]; };
  DecB dfb[2];
  for (int d = 0; d < 2; ++d) {
    dfb[d].We = tobf(df[d].We, (size_t)H * IN);
    dfb[d].Wa = tobf(df[d].Wa, (size_t)ML * 2 * H);
    dfb[d].Wc = tobf(df[d].Wc, (size_t)IN * 2 * H);
    for (int l = 0; l < 3; ++l) {
      dfb[d].l[l].Wih = tobf(df[d].l[l].Wih, (size_t)4 * H * (l ? H : IN));
      dfb[d].l[l].Whh = tobf(df[d].l[l].Whh, (size_t)4 * H * H);
    }
    for (int m = 0; m < 4; ++m)
      dfb[d].mW[m] = tobf(df[d].mW[m], (size_t)(m == 3 ? IN : H) * H);
  }

  // activations / state
  uint16_t* xbf   = (uint16_t*)alloc((size_t)B * IN * 2);
  uint16_t* t1bf  = (uint16_t*)alloc((size_t)B * H * 2);
  uint16_t* x2bf  = (uint16_t*)alloc((size_t)B * IN * 2);
  float*    G     = (float*)alloc((size_t)B * 4 * H * 4);
  uint16_t* ehbf[3]; float* ec[3];
  uint16_t* dhbf[3]; float* dc[3];
  for (int l = 0; l < 3; ++l) {
    ehbf[l] = (uint16_t*)alloc((size_t)B * H * 2);
    ec[l]   = (float*)alloc((size_t)B * H * 4);
    dhbf[l] = (uint16_t*)alloc((size_t)B * H * 2);
    dc[l]   = (float*)alloc((size_t)B * H * 4);
  }
  uint16_t* encbt = (uint16_t*)alloc((size_t)B * ML * H * 2);   // [B,ML,H]
  uint16_t* ybf   = (uint16_t*)alloc((size_t)B * IN * 2);
  uint16_t* embbf = (uint16_t*)alloc((size_t)B * H * 2);
  float*    awl   = (float*)alloc((size_t)B * ML * 4);
  float*    aw    = (float*)alloc((size_t)B * ML * 4);
  uint16_t* appbf = (uint16_t*)alloc((size_t)B * H * 2);
  float*    xinf  = (float*)alloc((size_t)B * IN * 4);
  uint16_t* xinbf = (uint16_t*)alloc((size_t)B * IN * 2);
  uint16_t* m1bf  = (uint16_t*)alloc((size_t)B * H * 2);
  uint16_t* m2bf  = (uint16_t*)alloc((size_t)B * H * 2);
  uint16_t* m3bf  = (uint16_t*)alloc((size_t)B * H * 2);

  auto gemm = [&](const uint16_t* X, int ldx, const uint16_t* W, int ldw,
                  const float* bias, float* Y, int ldy, int csy,
                  uint16_t* Ybf, int ldybf, int M, int N, int K, int flags) {
    const int blocks = (M / 64) * (N / 128);   // exact for all shapes here
    gemm_bf16_wmma_kernel<<<blocks, 256, 0, stream>>>(
        X, ldx, W, ldw, bias, Y, ldy, csy, Ybf, ldybf, M, N, K, flags);
  };

  auto lstm_stack = [&](const uint16_t* inp, int Kin, const LWb* wb,
                        const LWf* wf, uint16_t* const* hbf, float* const* cst,
                        uint16_t* encslot) {
    const uint16_t* curX = inp; int K0 = Kin;
    for (int l = 0; l < 3; ++l) {
      gemm(curX, K0, wb[l].Wih, K0, wf[l].bih, G, 4 * H, 1, nullptr, 0, B, 4 * H, K0, 0);
      gemm(hbf[l], H, wb[l].Whh, H, wf[l].bhh, G, 4 * H, 1, nullptr, 0, B, 4 * H, H, 1);
      lstm_gates_kernel<<<(B * H + 255) / 256, 256, 0, stream>>>(
          G, cst[l], hbf[l], (l == 2) ? encslot : nullptr, H, ML, B * H);
      curX = hbf[l]; K0 = H;
    }
  };

  // zero LSTM state
  for (int l = 0; l < 3; ++l) {
    fill_u32_kernel<<<(B * H / 2 + 255) / 256, 256, 0, stream>>>((uint32_t*)ehbf[l], B * H / 2);
    fill_u32_kernel<<<(B * H + 255) / 256, 256, 0, stream>>>((uint32_t*)ec[l], B * H);
  }

  // -------- encoder: MLP + 3-layer LSTM scan, attention bank into encbt -----
  for (int t = 0; t < S; ++t) {
    cvt_f32_bf16_kernel<<<(B * IN + 255) / 256, 256, 0, stream>>>(
        input_data + (size_t)t * B * IN, xbf, B * IN);
    gemm(xbf, IN, encW1b, IN, encb1, nullptr, 0, 0, t1bf, H, B, H, IN, 2);   // ReLU
    gemm(t1bf, H, encW2b, H, encb2, nullptr, 0, 0, x2bf, IN, B, IN, H, 2);   // ReLU
    lstm_stack(x2bf, IN, enclb, encl, ehbf, ec, encbt + (size_t)t * H);
  }

  // -------- two decoders (fwd, bwd), each seeded from encoder final state ---
  for (int d = 0; d < 2; ++d) {
    for (int l = 0; l < 3; ++l) {
      hipMemcpyAsync(dhbf[l], ehbf[l], (size_t)B * H * 2, hipMemcpyDeviceToDevice, stream);
      hipMemcpyAsync(dc[l], ec[l], (size_t)B * H * 4, hipMemcpyDeviceToDevice, stream);
    }
    cvt_f32_bf16_kernel<<<(B * IN + 255) / 256, 256, 0, stream>>>(
        input_data + (size_t)(S - 1) * B * IN, ybf, B * IN);
    float* outd = (float*)d_out + (size_t)d * B * IN * S;   // [B,IN,T], T fastest
    const DecB& Wb = dfb[d]; const DecF& Wf = df[d];

    for (int t = 0; t < S; ++t) {
      // emb = y @ We^T + be
      gemm(ybf, IN, Wb.We, IN, Wf.be, nullptr, 0, 0, embbf, H, B, H, IN, 0);
      // attention logits = [emb, h0] @ Wa^T + ba   (two GEMMs, ldw = 2H)
      gemm(embbf, H, Wb.Wa, 2 * H, Wf.ba, awl, ML, 1, nullptr, 0, B, ML, H, 0);
      gemm(dhbf[0], H, Wb.Wa + H, 2 * H, nullptr, awl, ML, 1, nullptr, 0, B, ML, H, 1);
      softmax128_kernel<<<B / 8, 256, 0, stream>>>(awl, aw, B);
      attn_ctx_kernel<<<(B * H + 255) / 256, 256, 0, stream>>>(aw, encbt, appbf, H, ML, B * H);
      // xin = [emb, app] @ Wc^T + bc
      gemm(embbf, H, Wb.Wc, 2 * H, Wf.bc, xinf, IN, 1, nullptr, 0, B, IN, H, 0);
      gemm(appbf, H, Wb.Wc + H, 2 * H, nullptr, xinf, IN, 1, xinbf, IN, B, IN, H, 1);
      lstm_stack(xinbf, IN, Wb.l, Wf.l, dhbf, dc, nullptr);
      // output MLP; final GEMM writes f32 straight into d_out[b,i,t] and a
      // bf16 copy that becomes the next step's input.
      gemm(dhbf[2], H, Wb.mW[0], H, Wf.mb[0], nullptr, 0, 0, m1bf, H, B, H, H, 2);
      gemm(m1bf, H, Wb.mW[1], H, Wf.mb[1], nullptr, 0, 0, m2bf, H, B, H, H, 2);
      gemm(m2bf, H, Wb.mW[2], H, Wf.mb[2], nullptr, 0, 0, m3bf, H, B, H, H, 2);
      gemm(m3bf, H, Wb.mW[3], H, Wf.mb[3], outd + t, IN * S, S, ybf, IN, B, IN, H, 0);
    }
  }
}
